// Sparsify_hypercol_local_modular_74775380623611
// MI455X (gfx1250) — matrix operand order repaired and checked by reading, not compiled, not run on hardware
//
#include <hip/hip_runtime.h>
#include <hip/hip_fp16.h>

typedef __attribute__((ext_vector_type(16))) _Float16 v16h;
typedef __attribute__((ext_vector_type(8)))  float    v8f;

#define N_BATCH 64
#define CH      128
#define RES     64
#define L       256
#define KTOP    25
#define NB      16
#define OC      13
#define INDIM   3456
#define HID     3456
#define OUTDIM  3328
#define ROWS    1024      // NB * N_BATCH
#define ASTRIDE 40        // LDS row stride in halves (80 B: 16B-aligned, spreads banks)

// Flip to 0 if the assembler rejects the async-to-LDS mnemonics.
#define USE_ASYNC_STAGING 1

union F16Frag { v16h v; uint4 q[2]; };

#if USE_ASYNC_STAGING
// Direct global -> LDS copy, tracked by ASYNCcnt (CDNA5 ISA 10.x / 15.18.3).
__device__ __forceinline__ void async_copy_b128(unsigned lds_addr, const _Float16* gaddr) {
    asm volatile("global_load_async_to_lds_b128 %0, %1, off"
                 :: "v"(lds_addr), "v"(gaddr)
                 : "memory");
}
__device__ __forceinline__ void wait_async0() {
    asm volatile("s_wait_asynccnt 0x0" ::: "memory");
}
__device__ __forceinline__ unsigned lds_addr_of(const void* p) {
    // generic LDS pointer: low 32 bits are the LDS byte offset
    return (unsigned)(size_t)p;
}
#endif

// ---------------------------------------------------------------------------
// fp32 -> f16 conversion WITH transpose: W[Kdim][Ndim] -> Wt[Ndim][Kdim].
// LDS-tiled 64x64 so both global streams are coalesced.  One pass over the
// weights; lets the GEMM stage B tiles with contiguous b128 like A.
// ---------------------------------------------------------------------------
__global__ __launch_bounds__(256) void cvt_transpose_f32_to_f16(
    const float* __restrict__ W, _Float16* __restrict__ Wt, int Kdim, int Ndim)
{
    __shared__ _Float16 tile[64][65];
    const int t  = threadIdx.x;
    const int n0 = blockIdx.x * 64;
    const int k0 = blockIdx.y * 64;
    for (int i = t; i < 64 * 64; i += 256) {
        int kk = i >> 6, nn = i & 63;
        tile[kk][nn] = (_Float16)W[(size_t)(k0 + kk) * Ndim + n0 + nn];
    }
    __syncthreads();
    for (int i = t; i < 64 * 64; i += 256) {
        int nn = i >> 6, kk = i & 63;
        Wt[(size_t)(n0 + nn) * Kdim + k0 + kk] = tile[kk][nn];
    }
}

// ---------------------------------------------------------------------------
// Per-(block g, batch b): score dot, exp, top-25 (tie -> lower index, like
// lax.top_k), sorted-index gather, f16 feat row build (pos-major, ch-minor,
// then 0/1 mask tail).  One 256-thread workgroup per row; thread == position.
// ---------------------------------------------------------------------------
__global__ __launch_bounds__(256) void select_feat_kernel(
    const float* __restrict__ x,        // [64][128][64][64]
    const float* __restrict__ conv_w,   // [16][128]
    _Float16* __restrict__ feat)        // [1024][3456] f16
{
    __shared__ float sP[256];
    __shared__ float sVal[256];
    __shared__ int   sIdx[256];
    __shared__ float sMask[256];
    __shared__ int   sTop[KTOP];

    const int t   = threadIdx.x;
    const int row = blockIdx.x;        // g*64 + b
    const int g   = row >> 6;
    const int b   = row & 63;
    const int gr  = g >> 2, gc = g & 3;
    const int lh  = t >> 4, lw = t & 15;
    const int h   = gr * 16 + lh, w = gc * 16 + lw;

    const size_t xbase = (size_t)b * CH * (RES * RES) + (size_t)h * RES + w;
    float s = 0.f;
#pragma unroll 8
    for (int c = 0; c < CH; ++c)
        s += x[xbase + (size_t)c * (RES * RES)] * conv_w[g * CH + c];

    sP[t]    = expf(s);   // ordering of e == ordering of p; p never enters output values
    sMask[t] = 0.f;
    __syncthreads();

    for (int k = 0; k < KTOP; ++k) {
        sVal[t] = sP[t]; sIdx[t] = t;
        __syncthreads();
        for (int stride = 128; stride > 0; stride >>= 1) {
            if (t < stride) {
                float v2 = sVal[t + stride]; int i2 = sIdx[t + stride];
                if (v2 > sVal[t] || (v2 == sVal[t] && i2 < sIdx[t])) {
                    sVal[t] = v2; sIdx[t] = i2;
                }
            }
            __syncthreads();
        }
        if (t == 0) {
            int bi = sIdx[0];
            sTop[k] = bi; sP[bi] = -1.f; sMask[bi] = 1.f;
        }
        __syncthreads();
    }

    if (t == 0) {   // sort the 25 indices ascending (spatial order)
        for (int i = 1; i < KTOP; ++i) {
            int key = sTop[i]; int j = i - 1;
            while (j >= 0 && sTop[j] > key) { sTop[j + 1] = sTop[j]; --j; }
            sTop[j + 1] = key;
        }
    }
    __syncthreads();

    _Float16* fr = feat + (size_t)row * INDIM;
    for (int i = t; i < KTOP * CH; i += 256) {
        int k  = i >> 7;            // position slot
        int c  = i & 127;           // channel
        int ls = sTop[k];
        int hh = gr * 16 + (ls >> 4), ww = gc * 16 + (ls & 15);
        float v = x[((size_t)b * CH + c) * (RES * RES) + (size_t)hh * RES + ww];
        fr[i] = (_Float16)v;
    }
    fr[KTOP * CH + t] = (_Float16)sMask[t];   // exact 0/1 straight-through values
}

// ---------------------------------------------------------------------------
// Double-buffered f16 WMMA GEMM:  out = act(A[1024xK] * Bt^T + bias)
//   A  row-major [M][K];  Bt pre-transposed weights [N][K].
// Block tile 128x128, 8 waves, wave tile 32x64 (2x4 WMMA accumulators).
// Staging: GLOBAL_LOAD_ASYNC_TO_LDS_B128 into the alternate LDS buffer while
// WMMAs consume the current one; s_wait_asynccnt 0 + one barrier per step.
// All LDS traffic is aligned b128 (row stride 80 B).
// ---------------------------------------------------------------------------
__global__ __launch_bounds__(256) void gemm_wmma(
    const _Float16* __restrict__ A,     // [1024][Kdim]
    const _Float16* __restrict__ Bt,    // [Ndim][Kdim]
    const float*    __restrict__ bias,  // [Ndim]
    _Float16*       __restrict__ outH,  // f16 output or nullptr
    float*          __restrict__ outF,  // f32 output or nullptr
    int Kdim, int Ndim, int applyRelu)
{
    __shared__ _Float16 aT[2][128 * ASTRIDE];
    __shared__ _Float16 bT[2][128 * ASTRIDE];

    const int t      = threadIdx.x;
    const int Nbase  = blockIdx.x * 128;
    const int Mbase  = blockIdx.y * 128;
    const int lane   = t & 31;
    const int wv     = t >> 5;
    const int lanelo = lane & 15;
    const int khalf  = lane >> 4;        // 0: lanes 0-15, 1: lanes 16-31
    const int Moff   = (wv & 3) * 32;    // 4 waves along M
    const int Noff   = (wv >> 2) * 64;   // 2 waves along N

    v8f acc[2][4];
#pragma unroll
    for (int mi = 0; mi < 2; ++mi)
#pragma unroll
        for (int ni = 0; ni < 4; ++ni)
#pragma unroll
            for (int r = 0; r < 8; ++r) acc[mi][ni][r] = 0.f;

    // staging pattern: 128 rows x 32 halves, 2x b128 per thread per matrix
    const int am   = t >> 1;
    const int aseg = (t & 1) * 16;
    const _Float16* gA = A  + (size_t)(Mbase + am) * Kdim + aseg;
    const _Float16* gB = Bt + (size_t)(Nbase + am) * Kdim + aseg;

#if USE_ASYNC_STAGING
    unsigned ldsA[2], ldsB[2];
#pragma unroll
    for (int bi = 0; bi < 2; ++bi) {
        ldsA[bi] = lds_addr_of(&aT[bi][am * ASTRIDE + aseg]);
        ldsB[bi] = lds_addr_of(&bT[bi][am * ASTRIDE + aseg]);
    }
#define STAGE_TILE(k0_, bi_)                                 \
    do {                                                     \
        async_copy_b128(ldsA[bi_],      gA + (k0_));         \
        async_copy_b128(ldsA[bi_] + 16, gA + (k0_) + 8);     \
        async_copy_b128(ldsB[bi_],      gB + (k0_));         \
        async_copy_b128(ldsB[bi_] + 16, gB + (k0_) + 8);     \
    } while (0)

    STAGE_TILE(0, 0);
    wait_async0();
    __syncthreads();
#else
    {
        uint4 ra0 = *(const uint4*)gA, ra1 = *(const uint4*)(gA + 8);
        uint4 rb0 = *(const uint4*)gB, rb1 = *(const uint4*)(gB + 8);
        *(uint4*)&aT[0][am * ASTRIDE + aseg]     = ra0;
        *(uint4*)&aT[0][am * ASTRIDE + aseg + 8] = ra1;
        *(uint4*)&bT[0][am * ASTRIDE + aseg]     = rb0;
        *(uint4*)&bT[0][am * ASTRIDE + aseg + 8] = rb1;
    }
    __syncthreads();
#endif

    int buf = 0;
    for (int k0 = 0; k0 < Kdim; k0 += 32) {
        const bool more = (k0 + 32) < Kdim;

#if USE_ASYNC_STAGING
        // issue next tile into the idle buffer; overlaps with the WMMAs below
        if (more) STAGE_TILE(k0 + 32, buf ^ 1);
#else
        uint4 ra0, ra1, rb0, rb1;
        if (more) {
            const _Float16* pa = gA + k0 + 32;
            const _Float16* pb = gB + k0 + 32;
            ra0 = *(const uint4*)pa; ra1 = *(const uint4*)(pa + 8);
            rb0 = *(const uint4*)pb; rb1 = *(const uint4*)(pb + 8);
            __builtin_prefetch(pa + 32, 0, 1);
            __builtin_prefetch(pb + 32, 0, 1);
        }
#endif

        // ---- fragments (CDNA5 ISA 7.12.2 layouts) ----
        F16Frag af[2], bf[4];
#pragma unroll
        for (int mi = 0; mi < 2; ++mi) {
            // A 16x32 f16: halves 0-7 = K[khalf*8..], halves 8-15 = K[16+khalf*8..]
            const _Float16* ap = &aT[buf][(Moff + mi * 16 + lanelo) * ASTRIDE];
            af[mi].q[0] = *(const uint4*)(ap + khalf * 8);
            af[mi].q[1] = *(const uint4*)(ap + 16 + khalf * 8);
        }
#pragma unroll
        for (int ni = 0; ni < 4; ++ni) {
            // B 32x16 f16: halves 0-7 = K[khalf*8..], halves 8-15 = K[16+khalf*8..]
            const _Float16* bp = &bT[buf][(Noff + ni * 16 + lanelo) * ASTRIDE];
            bf[ni].q[0] = *(const uint4*)(bp + khalf * 8);
            bf[ni].q[1] = *(const uint4*)(bp + 16 + khalf * 8);
        }

#pragma unroll
        for (int mi = 0; mi < 2; ++mi)
#pragma unroll
            for (int ni = 0; ni < 4; ++ni)
                acc[mi][ni] = __builtin_amdgcn_wmma_f32_16x16x32_f16(
                    false, af[mi].v, false, bf[ni].v,
                    (short)0, acc[mi][ni], false, false);

        if (more) {
#if USE_ASYNC_STAGING
            wait_async0();      // this wave's async copies landed in LDS
            __syncthreads();    // every wave's copies landed; reads of old buf done
#else
            __syncthreads();    // everyone done reading buf^1 (from 2 iters ago)
            *(uint4*)&aT[buf ^ 1][am * ASTRIDE + aseg]     = ra0;
            *(uint4*)&aT[buf ^ 1][am * ASTRIDE + aseg + 8] = ra1;
            *(uint4*)&bT[buf ^ 1][am * ASTRIDE + aseg]     = rb0;
            *(uint4*)&bT[buf ^ 1][am * ASTRIDE + aseg + 8] = rb1;
            __syncthreads();
#endif
            buf ^= 1;
        }
    }

    // ---- epilogue: bias (+ReLU), f16 or f32 store ----
    // C/D layout: lanes 0-15: N=lane, M=r; lanes 16-31: N=lane-16, M=8+r.
#pragma unroll
    for (int mi = 0; mi < 2; ++mi) {
#pragma unroll
        for (int ni = 0; ni < 4; ++ni) {
            const int Ncol = Nbase + Noff + ni * 16 + lanelo;
            const float bb = bias[Ncol];
#pragma unroll
            for (int r = 0; r < 8; ++r) {
                const int Mrow = Mbase + Moff + mi * 16 + r + khalf * 8;
                float v = acc[mi][ni][r] + bb;
                if (applyRelu) v = fmaxf(v, 0.f);
                const size_t o = (size_t)Mrow * Ndim + Ncol;
                if (outH) outH[o] = (_Float16)v;
                else      outF[o] = v;
            }
        }
    }
}

// ---------------------------------------------------------------------------
// Reassemble rec[1024][3328] into the spatial grid and apply the OC->CH 1x1
// conv.  One block covers 256 consecutive (h,w) of one batch image; stores
// are coalesced along w.
// ---------------------------------------------------------------------------
__global__ __launch_bounds__(256) void out_conv_kernel(
    const float* __restrict__ rec,    // [1024][3328]  (row = g*64+b, col = o*256+l)
    const float* __restrict__ out_w,  // [128][13]
    float* __restrict__ out)          // [64][128][64][64]
{
    __shared__ float sw[CH * OC];
    const int t = threadIdx.x;
    for (int i = t; i < CH * OC; i += 256) sw[i] = out_w[i];

    const int b   = blockIdx.x >> 4;
    const int seg = blockIdx.x & 15;
    const int p   = seg * 256 + t;           // hw index in [0,4096)
    const int h   = p >> 6, w = p & 63;
    const int g   = (h >> 4) * 4 + (w >> 4);
    const int l   = (h & 15) * 16 + (w & 15);

    const size_t rrow = (size_t)(g * 64 + b) * OUTDIM;
    float gv[OC];
#pragma unroll
    for (int o = 0; o < OC; ++o) gv[o] = rec[rrow + o * L + l];
    __syncthreads();

    float* ob = out + (size_t)b * CH * (RES * RES) + p;
#pragma unroll 4
    for (int c = 0; c < CH; ++c) {
        float a0 = 0.f;
#pragma unroll
        for (int o = 0; o < OC; ++o) a0 += gv[o] * sw[c * OC + o];
        ob[(size_t)c * (RES * RES)] = a0;
    }
}

// ---------------------------------------------------------------------------
extern "C" void kernel_launch(void* const* d_in, const int* in_sizes, int n_in,
                              void* d_out, int out_size, void* d_ws, size_t ws_size,
                              hipStream_t stream) {
    (void)in_sizes; (void)n_in; (void)out_size; (void)ws_size;
    const float* x      = (const float*)d_in[0];
    const float* conv_w = (const float*)d_in[1];
    const float* W1     = (const float*)d_in[2];
    const float* b1     = (const float*)d_in[3];
    const float* W2     = (const float*)d_in[4];
    const float* b2     = (const float*)d_in[5];
    const float* out_w  = (const float*)d_in[6];
    // d_in[7] = tau, unused by the forward pass
    float* out = (float*)d_out;

    char* ws = (char*)d_ws;
    size_t off = 0;
    _Float16* W1t   = (_Float16*)(ws + off); off += (size_t)INDIM * HID    * 2;  // [HID][INDIM]
    _Float16* W2t   = (_Float16*)(ws + off); off += (size_t)HID   * OUTDIM * 2;  // [OUTDIM][HID]
    _Float16* featH = (_Float16*)(ws + off); off += (size_t)ROWS  * INDIM  * 2;
    _Float16* hH    = (_Float16*)(ws + off); off += (size_t)ROWS  * HID    * 2;
    float*    rec   = (float*)   (ws + off); off += (size_t)ROWS  * OUTDIM * 4;

    // one-pass f16 conversion with transpose (weights become [N][K])
    cvt_transpose_f32_to_f16<<<dim3(HID / 64,    INDIM / 64), 256, 0, stream>>>(
        W1, W1t, INDIM, HID);
    cvt_transpose_f32_to_f16<<<dim3(OUTDIM / 64, HID / 64),   256, 0, stream>>>(
        W2, W2t, HID, OUTDIM);

    select_feat_kernel<<<ROWS, 256, 0, stream>>>(x, conv_w, featH);

    gemm_wmma<<<dim3(HID / 128, ROWS / 128), 256, 0, stream>>>(
        featH, W1t, b1, hH, nullptr, INDIM, HID, 1);
    gemm_wmma<<<dim3(OUTDIM / 128, ROWS / 128), 256, 0, stream>>>(
        hH, W2t, b2, nullptr, rec, HID, OUTDIM, 0);

    out_conv_kernel<<<N_BATCH * 16, 256, 0, stream>>>(rec, out_w, out);
}
